// ContrastLoss_32959579030314
// MI455X (gfx1250) — matrix operand order repaired and checked by reading, not compiled, not run on hardware
//
#include <hip/hip_runtime.h>

typedef __bf16 bf16_t;
typedef __attribute__((ext_vector_type(16))) __bf16 v16bf;
typedef __attribute__((ext_vector_type(8)))  float  v8f;
typedef __attribute__((ext_vector_type(4)))  float  f32x4;

#define TEMP_INV   5.0f      // 1 / 0.2
#define IOU_THRES  0.4f
#define PER_IMG    256
#define DIM        1024
#define NSCALE     4
#define NBATCH     128
#define HALVES     2         // blocks per image

// 256 blocks: block = (image b) * 2 + (half h). Each block covers 128
// proposals (8 waves x one 16-row tile) of the per-image
// [256 x 1024] x [1024 x 4(->16 padded)] GEMM using v_wmma_f32_16x16x32_bf16,
// fusing row sum-of-squares into the same (non-temporal) global sweep.
__global__ __launch_bounds__(256) void contrast_main(
    const float* __restrict__ box,   // [B*P, D]
    const float* __restrict__ crop,  // [K, B, D]
    const float* __restrict__ ious,  // [B*P]
    float* __restrict__ ws)          // partial sums, see layout below
{
    // ws layout: wsPos[4][128][2] | wsNeg[4][128][2] | wsCnt[128][2]
    float* wsPos = ws;
    float* wsNeg = ws + NSCALE * NBATCH * HALVES;
    float* wsCnt = ws + 2 * NSCALE * NBATCH * HALVES;

    const int b    = blockIdx.x >> 1;
    const int h    = blockIdx.x & 1;       // which half of the image's rows
    const int t    = threadIdx.x;
    const int wave = t >> 5;
    const int lane = t & 31;
    const int half = lane >> 4;            // lane half within the wave
    const int col  = lane & 15;            // A row within tile / B,C column

    __shared__ __align__(32) bf16_t zlds[NSCALE][DIM];  // crop feats as bf16
    __shared__ float zpart[256];
    __shared__ float posf[128];            // pos mask for this half's rows
    __shared__ float znq[NSCALE];
    __shared__ float wpos[8][NSCALE];
    __shared__ float wneg[8][NSCALE];
    __shared__ int   cnt;

    if (t == 0) cnt = 0;
    __syncthreads();

    // ---- Stage crop features (f32 -> bf16) into LDS; pos mask; count ----
    {
        const int k  = t >> 6;             // 0..3
        const int d0 = (t & 63) << 4;      // 0..1008 step 16
        const float* src = crop + ((size_t)(k * NBATCH + b)) * DIM + d0;
        f32x4 ssv = {0.f, 0.f, 0.f, 0.f};
        #pragma unroll
        for (int i = 0; i < 4; ++i) {
            f32x4 f = reinterpret_cast<const f32x4*>(src)[i];
            ssv += f * f;
            zlds[k][d0 + 4 * i + 0] = (bf16_t)f.x;
            zlds[k][d0 + 4 * i + 1] = (bf16_t)f.y;
            zlds[k][d0 + 4 * i + 2] = (bf16_t)f.z;
            zlds[k][d0 + 4 * i + 3] = (bf16_t)f.w;
        }
        zpart[t] = (ssv.x + ssv.y) + (ssv.z + ssv.w);

        if (t < 128) {
            float iou = ious[(size_t)b * PER_IMG + h * 128 + t];
            int pos   = (iou >= IOU_THRES) ? 1 : 0;
            posf[t]   = (float)pos;
            if (pos) atomicAdd(&cnt, 1);   // integer atomic: deterministic
        }
    }
    __syncthreads();

    // Deterministic fixed-order reduction of z sum-of-squares
    if (t < NSCALE) {
        float s = 0.f;
        for (int i = 0; i < 64; ++i) s += zpart[t * 64 + i];
        znq[t] = s;
    }
    __syncthreads();

    // Columns >= 4 are replicas of real data; killed below via zinv == 0.
    const float zinv =
        (col < NSCALE) ? (1.0f / fmaxf(sqrtf(znq[col]), 1e-12f)) : 0.0f;

    // ---- One 16-row tile per wave ----
    const int tile = wave;                          // 0..7 within this half
    const float* arow =
        box + ((size_t)(b * PER_IMG + h * 128 + tile * 16 + col)) * DIM
            + half * 16;
    // B fragment source: column col (replicated for col>=4), K-half 'half'
    const v16bf* zcol =
        reinterpret_cast<const v16bf*>(&zlds[col & (NSCALE - 1)][half * 16]);

    v8f   c    = {};
    f32x4 ssqv = {0.f, 0.f, 0.f, 0.f};   // vector accumulator: pk_fma only

    #pragma unroll 4
    for (int kb = 0; kb < DIM; kb += 32) {
        // A 16x32 bf16 fragment: one contiguous 64B run -> NT b128 loads
        f32x4 f0 = __builtin_nontemporal_load(
                       reinterpret_cast<const f32x4*>(arow + kb + 0));
        f32x4 f1 = __builtin_nontemporal_load(
                       reinterpret_cast<const f32x4*>(arow + kb + 4));
        f32x4 f2 = __builtin_nontemporal_load(
                       reinterpret_cast<const f32x4*>(arow + kb + 8));
        f32x4 f3 = __builtin_nontemporal_load(
                       reinterpret_cast<const f32x4*>(arow + kb + 12));
        ssqv += f0 * f0;
        ssqv += f1 * f1;
        ssqv += f2 * f2;
        ssqv += f3 * f3;

        v16bf a;
        a[0]  = (bf16_t)f0.x; a[1]  = (bf16_t)f0.y;
        a[2]  = (bf16_t)f0.z; a[3]  = (bf16_t)f0.w;
        a[4]  = (bf16_t)f1.x; a[5]  = (bf16_t)f1.y;
        a[6]  = (bf16_t)f1.z; a[7]  = (bf16_t)f1.w;
        a[8]  = (bf16_t)f2.x; a[9]  = (bf16_t)f2.y;
        a[10] = (bf16_t)f2.z; a[11] = (bf16_t)f2.w;
        a[12] = (bf16_t)f3.x; a[13] = (bf16_t)f3.y;
        a[14] = (bf16_t)f3.z; a[15] = (bf16_t)f3.w;

        v16bf bb = zcol[kb >> 4];   // K = kb + half*16 + [0..15]

        c = __builtin_amdgcn_wmma_f32_16x16x32_bf16(
                false, a, false, bb, (short)0, c, false, false);
    }

    // Horizontal reduce (once), then combine lane pair for full row sumsq
    float ssq  = (ssqv.x + ssqv.y) + (ssqv.z + ssqv.w);
    float ssqf = ssq + __shfl_xor(ssq, 16, 32);

    float posS = 0.f, negS = 0.f;
    #pragma unroll
    for (int r = 0; r < 8; ++r) {
        int   row  = r + 8 * half;                 // C row for c[r]
        float srow = __shfl(ssqf, row, 32);        // lane 'row' holds it
        float rinv = 1.0f / fmaxf(sqrtf(srow), 1e-12f);
        float cosv = c[r] * rinv * zinv;           // cosine similarity
        float pf   = posf[tile * 16 + row];
        posS += cosv * pf;
        negS += cosv * (1.0f - pf);
    }

    // Combine lane pair (col, col+16); one lane per k stores wave partial
    posS += __shfl_xor(posS, 16, 32);
    negS += __shfl_xor(negS, 16, 32);
    if (lane < NSCALE) { wpos[wave][lane] = posS; wneg[wave][lane] = negS; }
    __syncthreads();

    // Deterministic fixed-order combine across waves -> global partials
    if (t < NSCALE) {
        float ps = 0.f, ns = 0.f;
        for (int w = 0; w < 8; ++w) { ps += wpos[w][t]; ns += wneg[w][t]; }
        wsPos[(t * NBATCH + b) * HALVES + h] = ps;
        wsNeg[(t * NBATCH + b) * HALVES + h] = ns;
    }
    if (t == 0) wsCnt[b * HALVES + h] = (float)cnt;
}

// Combine halves, per-(k,b) softplus, deterministic sum over b, min over k.
__global__ __launch_bounds__(128) void contrast_final(
    const float* __restrict__ ws, float* __restrict__ out)
{
    const float* wsPos = ws;
    const float* wsNeg = ws + NSCALE * NBATCH * HALVES;
    const float* wsCnt = ws + 2 * NSCALE * NBATCH * HALVES;

    __shared__ float spv[NSCALE][NBATCH];

    const int b = threadIdx.x;              // 0..127
    float cp = wsCnt[b * HALVES + 0] + wsCnt[b * HALVES + 1];
    float cn = (float)PER_IMG - cp;
    #pragma unroll
    for (int k = 0; k < NSCALE; ++k) {
        float ps = wsPos[(k * NBATCH + b) * HALVES + 0]
                 + wsPos[(k * NBATCH + b) * HALVES + 1];
        float ns = wsNeg[(k * NBATCH + b) * HALVES + 0]
                 + wsNeg[(k * NBATCH + b) * HALVES + 1];
        float sim_pos = -ps / cp;
        float sim_neg = -ns / cn;
        float x  = (sim_neg - sim_pos) * TEMP_INV;
        spv[k][b] = fmaxf(x, 0.f) + log1pf(expf(-fabsf(x)));   // softplus
    }
    __syncthreads();

    if (b == 0) {
        float mn = 3.402823466e38f;
        for (int k = 0; k < NSCALE; ++k) {
            float s = 0.f;
            for (int i = 0; i < NBATCH; ++i) s += spv[k][i];   // fixed order
            mn = fminf(mn, s);
        }
        out[0] = mn / (float)NBATCH;
    }
}

extern "C" void kernel_launch(void* const* d_in, const int* in_sizes, int n_in,
                              void* d_out, int out_size, void* d_ws, size_t ws_size,
                              hipStream_t stream) {
    const float* box  = (const float*)d_in[0];  // [32768, 1024] f32
    const float* crop = (const float*)d_in[1];  // [4, 128, 1024] f32
    // d_in[2] = batch_size scalar (== 128, baked into grid)
    const float* ious = (const float*)d_in[3];  // [32768] f32
    float* ws  = (float*)d_ws;                  // ~9 KB of partials
    float* out = (float*)d_out;                 // scalar f32

    contrast_main<<<NBATCH * HALVES, 256, 0, stream>>>(box, crop, ious, ws);
    contrast_final<<<1, 128, 0, stream>>>(ws, out);
}
